// TransformerEncoderLayer_13125420056970
// MI455X (gfx1250) — matrix operand order repaired and checked by reading, not compiled, not run on hardware
//
#include <hip/hip_runtime.h>
#include <hip/hip_bf16.h>
#include <cstdint>
#include <cstddef>

typedef __attribute__((ext_vector_type(16))) _Float16 v16h;
typedef __attribute__((ext_vector_type(8)))  float    v8f;
typedef __attribute__((ext_vector_type(4)))  _Float16 v4h;

#define TE_B  2
#define TE_S  2048
#define TE_D  1024
#define TE_H  16
#define TE_F  4096
#define TE_HD 64
#define BSROWS (TE_B * TE_S)   // 4096 rows

// ---------------------------------------------------------------------------
// CDNA5 async copy: global -> LDS, 16 bytes per lane, tracked by ASYNCcnt.
// LDS operand: wave-relative byte offset == low 32 bits of the generic pointer
// (ISA 10.2: LDS aperture address truncates to addr[31:0]).
// ---------------------------------------------------------------------------
__device__ __forceinline__ void async_ld_b128(const void* lds, const void* gaddr) {
  unsigned l = (unsigned)(uintptr_t)lds;
  uint64_t g = (uint64_t)(uintptr_t)gaddr;
  asm volatile("global_load_async_to_lds_b128 %0, %1, off" :: "v"(l), "v"(g) : "memory");
}
__device__ __forceinline__ void wait_async0() {
  asm volatile("s_wait_asynccnt 0x0" ::: "memory");
}

// ---------------------------------------------------------------------------
// WMMA fragment loaders (wave32, V_WMMA_F32_16X16X32_F16 layouts, ISA 7.12.2)
// ---------------------------------------------------------------------------
__device__ __forceinline__ v16h ld_a_frag(const _Float16* s, int stride, int r, int hg, int kb) {
  v16h a;
  const _Float16* p = s + r * stride + kb + hg * 8;
#pragma unroll
  for (int i = 0; i < 8; i++) a[i] = p[i];
#pragma unroll
  for (int i = 0; i < 8; i++) a[8 + i] = p[16 + i];
  return a;
}
__device__ __forceinline__ v16h ld_b_frag(const _Float16* s, int stride, int r, int hg, int kb) {
  v16h b;
  const _Float16* p = s + r * stride + kb + hg * 8;
#pragma unroll
  for (int i = 0; i < 8; i++) b[i] = p[i];
#pragma unroll
  for (int i = 0; i < 8; i++) b[8 + i] = p[16 + i];
  return b;
}
__device__ __forceinline__ v8f v8f_zero() {
  v8f z;
#pragma unroll
  for (int i = 0; i < 8; i++) z[i] = 0.0f;
  return z;
}

// ---------------------------------------------------------------------------
// Transpose + convert: in[K][N] f32 -> out[N][K] f16 (weights, once per launch)
// ---------------------------------------------------------------------------
__global__ __launch_bounds__(256) void transpose_cvt_kernel(const float* __restrict__ in,
                                                            _Float16* __restrict__ out,
                                                            int K, int N) {
  __shared__ float tile[32][33];
  const int k0 = blockIdx.y * 32, n0 = blockIdx.x * 32;
  const int tx = threadIdx.x & 31, ty = threadIdx.x >> 5;   // 8 rows of 32
#pragma unroll
  for (int i = 0; i < 4; i++)
    tile[ty + i * 8][tx] = in[(size_t)(k0 + ty + i * 8) * N + n0 + tx];
  __syncthreads();
#pragma unroll
  for (int i = 0; i < 4; i++)
    out[(size_t)(n0 + ty + i * 8) * K + k0 + tx] = (_Float16)tile[tx][ty + i * 8];
}

// ---------------------------------------------------------------------------
// LayerNorm over D=1024, one block per row, f32 in -> f16 out
// ---------------------------------------------------------------------------
__global__ __launch_bounds__(256) void layernorm_f16_kernel(const float* __restrict__ x,
                                                            const float* __restrict__ g,
                                                            const float* __restrict__ bt,
                                                            _Float16* __restrict__ y) {
  __shared__ float rs[256], rq[256];
  const int row = blockIdx.x, tid = threadIdx.x;
  const int c = tid * 4;
  float4 v = *(const float4*)(x + (size_t)row * TE_D + c);
  rs[tid] = v.x + v.y + v.z + v.w;
  rq[tid] = v.x * v.x + v.y * v.y + v.z * v.z + v.w * v.w;
  __syncthreads();
  for (int off = 128; off > 0; off >>= 1) {
    if (tid < off) { rs[tid] += rs[tid + off]; rq[tid] += rq[tid + off]; }
    __syncthreads();
  }
  const float mean = rs[0] * (1.0f / TE_D);
  const float var  = rq[0] * (1.0f / TE_D) - mean * mean;
  const float rstd = rsqrtf(var + 1e-5f);
  float4 gv = *(const float4*)(g + c);
  float4 bv = *(const float4*)(bt + c);
  v4h h;
  h[0] = (_Float16)((v.x - mean) * rstd * gv.x + bv.x);
  h[1] = (_Float16)((v.y - mean) * rstd * gv.y + bv.y);
  h[2] = (_Float16)((v.z - mean) * rstd * gv.z + bv.z);
  h[3] = (_Float16)((v.w - mean) * rstd * gv.w + bv.w);
  *(v4h*)(y + (size_t)row * TE_D + c) = h;
}

// ---------------------------------------------------------------------------
// WMMA GEMM: out[M,N] = A[M,K](f16) @ Wt[N,K](f16, pre-transposed) + bias
//            [+res] [relu].  Block tile 128x128xBK32, double-buffered LDS,
//            async global->LDS staging.  8 waves (4M x 2N), wave tile 32x64.
// ---------------------------------------------------------------------------
template <bool RELU, bool HAS_RES, bool OUT16>
__global__ __launch_bounds__(256) void gemm_wmma_kernel(
    const _Float16* __restrict__ A, const _Float16* __restrict__ Wt,
    const float* __restrict__ bias, const float* __restrict__ res,
    float* __restrict__ outF, _Float16* __restrict__ outH,
    int M, int N, int K) {
  constexpr int BK = 32;
  constexpr int LDA = 40;                       // padded stride (halves)
  __shared__ _Float16 As[2][128 * LDA];         // [row][k]
  __shared__ _Float16 Bs[2][128 * LDA];         // [n][k]

  const int tid = threadIdx.x;
  const int wid = tid >> 5, lane = tid & 31;
  const int hg = lane >> 4, r = lane & 15;
  const int m0 = blockIdx.y * 128;
  const int n0 = blockIdx.x * 128;
  const int wm = (wid & 3) * 32;
  const int wn = (wid >> 2) * 64;

  v8f acc[2][4];
#pragma unroll
  for (int mt = 0; mt < 2; mt++)
#pragma unroll
    for (int nt = 0; nt < 4; nt++) acc[mt][nt] = v8f_zero();

  auto stage = [&](int buf, int k0) {
#pragma unroll
    for (int l = 0; l < 2; l++) {
      int vI = tid + l * 256;
      int row = vI >> 2, col = (vI & 3) * 8;   // 4 x b128 per 32-half row
      async_ld_b128(&As[buf][row * LDA + col], A  + (size_t)(m0 + row) * K + k0 + col);
      async_ld_b128(&Bs[buf][row * LDA + col], Wt + (size_t)(n0 + row) * K + k0 + col);
    }
  };

  stage(0, 0);
  wait_async0();
  __syncthreads();

  int buf = 0;
  for (int k0 = 0; k0 < K; k0 += BK) {
    if (k0 + BK < K) stage(buf ^ 1, k0 + BK);   // overlap next tile with compute

    v16h af[2];
#pragma unroll
    for (int mt = 0; mt < 2; mt++) af[mt] = ld_a_frag(&As[buf][(wm + mt * 16) * LDA], LDA, r, hg, 0);
    v16h bf[4];
#pragma unroll
    for (int nt = 0; nt < 4; nt++) bf[nt] = ld_b_frag(&Bs[buf][(wn + nt * 16) * LDA], LDA, r, hg, 0);
#pragma unroll
    for (int mt = 0; mt < 2; mt++)
#pragma unroll
      for (int nt = 0; nt < 4; nt++)
        acc[mt][nt] = __builtin_amdgcn_wmma_f32_16x16x32_f16(
            false, af[mt], false, bf[nt], (short)0, acc[mt][nt], false, false);

    wait_async0();      // next tile landed (after compute -> usually free)
    __syncthreads();
    buf ^= 1;
  }

#pragma unroll
  for (int mt = 0; mt < 2; mt++) {
#pragma unroll
    for (int nt = 0; nt < 4; nt++) {
      const int colg = n0 + wn + nt * 16 + r;
      const float bv = bias[colg];
#pragma unroll
      for (int j = 0; j < 8; j++) {
        const int rowg = m0 + wm + mt * 16 + hg * 8 + j;
        const size_t idx = (size_t)rowg * N + colg;
        float val = acc[mt][nt][j] + bv;
        if (HAS_RES) val += res[idx];
        if (RELU) val = fmaxf(val, 0.0f);
        if (OUT16) outH[idx] = (_Float16)val;
        else       outF[idx] = val;
      }
    }
  }
}

// ---------------------------------------------------------------------------
// Flash attention: grid (S/128, B*H). 8 waves, each owns 16 query rows.
// q/k/v/ctx layout [B, S, H*HD] f16; head = column slice at h*HD.
// Q/K tiles staged with async global->LDS; V transposed through VGPRs.
// ---------------------------------------------------------------------------
__global__ __launch_bounds__(256) void attention_kernel(
    const _Float16* __restrict__ q, const _Float16* __restrict__ k,
    const _Float16* __restrict__ v, const int* __restrict__ mask,
    _Float16* __restrict__ ctx) {
  __shared__ _Float16 Qs[128 * 72];       // Q tile [128][64]+pad
  __shared__ _Float16 KVs[64 * 72];       // K tile [key][hd]  /  V^T [hd][key] (union)
  __shared__ _Float16 Ps[8 * 16 * 68];    // per-wave P (16x64)+pad
  __shared__ int Msk[64];

  const int tid = threadIdx.x;
  const int wid = tid >> 5, lane = tid & 31;
  const int hg = lane >> 4, r = lane & 15;
  const int qb = blockIdx.x;
  const int bh = blockIdx.y;
  const int b = bh >> 4, h = bh & 15;
  const size_t base = (size_t)b * TE_S * TE_D + (size_t)h * TE_HD;
  const int wm = wid * 16;

  // Stage Q tile (128 x 64) asynchronously
#pragma unroll
  for (int l = 0; l < 4; l++) {
    int vI = tid + l * 256;
    int row = vI >> 3, col = (vI & 7) * 8;
    async_ld_b128(&Qs[row * 72 + col], q + base + (size_t)(qb * 128 + row) * TE_D + col);
  }

  v8f oacc[4];
#pragma unroll
  for (int nt = 0; nt < 4; nt++) oacc[nt] = v8f_zero();
  float mstat[8], lstat[8];
#pragma unroll
  for (int j = 0; j < 8; j++) { mstat[j] = -1e30f; lstat[j] = 0.0f; }

  for (int kb = 0; kb < TE_S / 64; kb++) {
    // Stage K tile (64 x 64) asynchronously; mask via regular path
#pragma unroll
    for (int l = 0; l < 2; l++) {
      int vI = tid + l * 256;
      int row = vI >> 3, col = (vI & 7) * 8;
      async_ld_b128(&KVs[row * 72 + col], k + base + (size_t)(kb * 64 + row) * TE_D + col);
    }
    if (tid < 64) Msk[tid] = mask[b * TE_S + kb * 64 + tid];
    wait_async0();
    __syncthreads();

    // S = Q @ K^T  (16x64 per wave), contraction over HD=64 in 2 steps
    v8f sacc[4];
#pragma unroll
    for (int nt = 0; nt < 4; nt++) sacc[nt] = v8f_zero();
#pragma unroll
    for (int kk = 0; kk < 64; kk += 32) {
      v16h af = ld_a_frag(&Qs[wm * 72], 72, r, hg, kk);
#pragma unroll
      for (int nt = 0; nt < 4; nt++) {
        v16h bf = ld_b_frag(&KVs[nt * 16 * 72], 72, r, hg, kk);
        sacc[nt] = __builtin_amdgcn_wmma_f32_16x16x32_f16(
            false, af, false, bf, (short)0, sacc[nt], false, false);
      }
    }
    __syncthreads();  // all waves done reading K tile

    // Online softmax (scale 1/sqrt(64), mask==0 -> -1000 logit)
    float sv[4][8];
#pragma unroll
    for (int nt = 0; nt < 4; nt++) {
      const int mk = Msk[nt * 16 + r];
#pragma unroll
      for (int j = 0; j < 8; j++)
        sv[nt][j] = mk ? sacc[nt][j] * 0.125f : -1000.0f;
    }
    float pm[8];
#pragma unroll
    for (int j = 0; j < 8; j++)
      pm[j] = fmaxf(fmaxf(sv[0][j], sv[1][j]), fmaxf(sv[2][j], sv[3][j]));
#pragma unroll
    for (int off = 1; off < 16; off <<= 1)
#pragma unroll
      for (int j = 0; j < 8; j++) pm[j] = fmaxf(pm[j], __shfl_xor(pm[j], off, 32));
    float alpha[8];
#pragma unroll
    for (int j = 0; j < 8; j++) {
      float mn = fmaxf(mstat[j], pm[j]);
      alpha[j] = __expf(mstat[j] - mn);
      mstat[j] = mn;
    }
    float rsum[8];
#pragma unroll
    for (int j = 0; j < 8; j++) rsum[j] = 0.0f;
#pragma unroll
    for (int nt = 0; nt < 4; nt++)
#pragma unroll
      for (int j = 0; j < 8; j++) {
        float pval = __expf(sv[nt][j] - mstat[j]);
        rsum[j] += pval;
        Ps[wid * 1088 + (hg * 8 + j) * 68 + nt * 16 + r] = (_Float16)pval;
      }
#pragma unroll
    for (int off = 1; off < 16; off <<= 1)
#pragma unroll
      for (int j = 0; j < 8; j++) rsum[j] += __shfl_xor(rsum[j], off, 32);
#pragma unroll
    for (int j = 0; j < 8; j++) lstat[j] = lstat[j] * alpha[j] + rsum[j];
#pragma unroll
    for (int nt = 0; nt < 4; nt++)
#pragma unroll
      for (int j = 0; j < 8; j++) oacc[nt][j] *= alpha[j];

    // Stage V transposed into KVs: Vt[hd][key] (inherent transpose, via VGPRs)
#pragma unroll
    for (int l = 0; l < 2; l++) {
      int vI = tid + l * 256;
      int row = vI >> 3, col = (vI & 7) * 8;
      uint4 d = *(const uint4*)(v + base + (size_t)(kb * 64 + row) * TE_D + col);
      const _Float16* t = (const _Float16*)&d;
#pragma unroll
      for (int j = 0; j < 8; j++) KVs[(col + j) * 72 + row] = t[j];
    }
    __syncthreads();

    // O += P @ V  (contraction over 64 keys, 2 steps)
#pragma unroll
    for (int kk = 0; kk < 64; kk += 32) {
      v16h af = ld_a_frag(&Ps[wid * 1088], 68, r, hg, kk);
#pragma unroll
      for (int nt = 0; nt < 4; nt++) {
        v16h bf = ld_b_frag(&KVs[nt * 16 * 72], 72, r, hg, kk);
        oacc[nt] = __builtin_amdgcn_wmma_f32_16x16x32_f16(
            false, af, false, bf, (short)0, oacc[nt], false, false);
      }
    }
    __syncthreads();  // protect KVs/Msk before next iteration
  }

#pragma unroll
  for (int j = 0; j < 8; j++) {
    const float inv = 1.0f / lstat[j];
    const int rowq = qb * 128 + wm + hg * 8 + j;
#pragma unroll
    for (int nt = 0; nt < 4; nt++)
      ctx[base + (size_t)rowq * TE_D + nt * 16 + r] = (_Float16)(oacc[nt][j] * inv);
  }
}

// ---------------------------------------------------------------------------
// Host launcher
// ---------------------------------------------------------------------------
extern "C" void kernel_launch(void* const* d_in, const int* in_sizes, int n_in,
                              void* d_out, int out_size, void* d_ws, size_t ws_size,
                              hipStream_t stream) {
  (void)in_sizes; (void)n_in; (void)out_size; (void)ws_size;
  const float* x    = (const float*)d_in[0];
  const int*   msk  = (const int*)d_in[1];
  const float* Wq   = (const float*)d_in[2];
  const float* bq   = (const float*)d_in[3];
  const float* Wk   = (const float*)d_in[4];
  const float* bk   = (const float*)d_in[5];
  const float* Wv   = (const float*)d_in[6];
  const float* bv   = (const float*)d_in[7];
  const float* Wo   = (const float*)d_in[8];
  const float* bo   = (const float*)d_in[9];
  const float* g1   = (const float*)d_in[10];
  const float* be1  = (const float*)d_in[11];
  const float* g2   = (const float*)d_in[12];
  const float* be2  = (const float*)d_in[13];
  const float* W1   = (const float*)d_in[14];
  const float* bf1  = (const float*)d_in[15];
  const float* W2   = (const float*)d_in[16];
  const float* bf2  = (const float*)d_in[17];
  float* out = (float*)d_out;

  char* p = (char*)d_ws;
  auto alloc = [&](size_t bytes) -> char* {
    char* rp = p;
    p += (bytes + 255) & ~(size_t)255;
    return rp;
  };
  const size_t DD = (size_t)TE_D * TE_D;
  const size_t DF = (size_t)TE_D * TE_F;
  const size_t ND = (size_t)BSROWS * TE_D;
  const size_t NF = (size_t)BSROWS * TE_F;

  _Float16* wq16 = (_Float16*)alloc(DD * 2);   // [N][K] transposed f16
  _Float16* wk16 = (_Float16*)alloc(DD * 2);
  _Float16* wv16 = (_Float16*)alloc(DD * 2);
  _Float16* wo16 = (_Float16*)alloc(DD * 2);
  _Float16* w116 = (_Float16*)alloc(DF * 2);   // [F][D]
  _Float16* w216 = (_Float16*)alloc(DF * 2);   // [D][F]
  _Float16* y1h  = (_Float16*)alloc(ND * 2);
  _Float16* qh   = (_Float16*)alloc(ND * 2);
  _Float16* kh   = (_Float16*)alloc(ND * 2);
  _Float16* vh   = (_Float16*)alloc(ND * 2);
  _Float16* ctxh = (_Float16*)alloc(ND * 2);
  float*    y1r  = (float*)alloc(ND * 4);
  _Float16* y2h  = (_Float16*)alloc(ND * 2);
  _Float16* ffh  = (_Float16*)alloc(NF * 2);

  // 1) Weight transpose+convert: in[K][N] -> out[N][K] f16
  dim3 tDD(TE_D / 32, TE_D / 32);
  transpose_cvt_kernel<<<tDD, 256, 0, stream>>>(Wq, wq16, TE_D, TE_D);
  transpose_cvt_kernel<<<tDD, 256, 0, stream>>>(Wk, wk16, TE_D, TE_D);
  transpose_cvt_kernel<<<tDD, 256, 0, stream>>>(Wv, wv16, TE_D, TE_D);
  transpose_cvt_kernel<<<tDD, 256, 0, stream>>>(Wo, wo16, TE_D, TE_D);
  transpose_cvt_kernel<<<dim3(TE_F / 32, TE_D / 32), 256, 0, stream>>>(W1, w116, TE_D, TE_F);
  transpose_cvt_kernel<<<dim3(TE_D / 32, TE_F / 32), 256, 0, stream>>>(W2, w216, TE_F, TE_D);

  // 2) LN1
  layernorm_f16_kernel<<<BSROWS, 256, 0, stream>>>(x, g1, be1, y1h);

  // 3) QKV projections
  dim3 gDD(TE_D / 128, BSROWS / 128);
  gemm_wmma_kernel<false, false, true><<<gDD, 256, 0, stream>>>(
      y1h, wq16, bq, nullptr, nullptr, qh, BSROWS, TE_D, TE_D);
  gemm_wmma_kernel<false, false, true><<<gDD, 256, 0, stream>>>(
      y1h, wk16, bk, nullptr, nullptr, kh, BSROWS, TE_D, TE_D);
  gemm_wmma_kernel<false, false, true><<<gDD, 256, 0, stream>>>(
      y1h, wv16, bv, nullptr, nullptr, vh, BSROWS, TE_D, TE_D);

  // 4) Attention
  attention_kernel<<<dim3(TE_S / 128, TE_B * TE_H), 256, 0, stream>>>(qh, kh, vh, msk, ctxh);

  // 5) O projection + residual(x) -> y1r (f32)
  gemm_wmma_kernel<false, true, false><<<gDD, 256, 0, stream>>>(
      ctxh, wo16, bo, x, y1r, nullptr, BSROWS, TE_D, TE_D);

  // 6) LN2
  layernorm_f16_kernel<<<BSROWS, 256, 0, stream>>>(y1r, g2, be2, y2h);

  // 7) FFN1 with ReLU -> f16
  gemm_wmma_kernel<true, false, true><<<dim3(TE_F / 128, BSROWS / 128), 256, 0, stream>>>(
      y2h, w116, bf1, nullptr, nullptr, ffh, BSROWS, TE_F, TE_D);

  // 8) FFN2 + residual(y1r) -> d_out (f32)
  gemm_wmma_kernel<false, true, false><<<gDD, 256, 0, stream>>>(
      ffh, w216, bf2, y1r, out, nullptr, BSROWS, TE_D, TE_F);
}
